// LSTMClassifier_85590108275267
// MI455X (gfx1250) — compile-verified, matrix operand order checked by reading, not run on hardware
//
#include <hip/hip_runtime.h>

#define BATCH 4096
#define T_STEPS 128
#define I_DIM 32
#define H_DIM 64
#define G_DIM 256   // 4*H
#define BT 16       // batch tile per workgroup
#define NTHREADS 256

typedef __attribute__((ext_vector_type(16))) _Float16 v16h;
typedef __attribute__((ext_vector_type(8)))  _Float16 v8h;
typedef __attribute__((ext_vector_type(8)))  float    v8f;

__device__ __forceinline__ float sigf(float v) { return 1.0f / (1.0f + __expf(-v)); }
__device__ __forceinline__ float tanh_fast(float v) {
  float e = __expf(-2.0f * v);
  return (1.0f - e) / (1.0f + e);
}

// A-matrix fragment (16xK tile, f16) from an LDS row-major [16][lda] buffer.
// Per ISA 7.12.2 layout: lanes 0-15 rows M=0..15 hold K=[0..7]+[16..23] (k0-relative),
// lanes 16-31 hold K=[8..15]+[24..31]. Each half is 8 contiguous halves -> ds_load_b128.
__device__ __forceinline__ v16h load_A_lds(const _Float16* base, int lda, int k0) {
  const int lane = threadIdx.x & 31;
  const int m = lane & 15, hi = lane >> 4;
  const _Float16* p = base + m * lda + k0 + 8 * hi;
  v8h lo = *(const v8h*)(p);
  v8h hv = *(const v8h*)(p + 16);
  v16h a;
#pragma unroll
  for (int e = 0; e < 8; ++e) { a[e] = lo[e]; a[8 + e] = hv[e]; }
  return a;
}

// B-matrix fragment (Kx16 tile) built directly from global f32 weights W[ng][ldw]
// (B[k][n] = W[n0+n][k0+k]); converted to f16 once, lives in VGPRs for the whole loop.
__device__ __forceinline__ v16h load_B_gmem_f32(const float* __restrict__ W, int ldw,
                                                int n0, int k0) {
  const int lane = threadIdx.x & 31;
  const int n = lane & 15, hi = lane >> 4;
  const float* p = W + (size_t)(n0 + n) * ldw + k0 + 8 * hi;
  v16h b;
#pragma unroll
  for (int e = 0; e < 8; ++e) { b[e] = (_Float16)p[e]; b[8 + e] = (_Float16)p[e + 16]; }
  return b;
}

#define WMMA_F16(A, Bm, C) \
  __builtin_amdgcn_wmma_f32_16x16x32_f16(false, (A), false, (Bm), (short)0, (C), false, false)

__global__ __launch_bounds__(NTHREADS, 1)
void lstm2_fused_wmma_kernel(const float* __restrict__ x,
                             const float* __restrict__ Wih0, const float* __restrict__ Whh0,
                             const float* __restrict__ bih0, const float* __restrict__ bhh0,
                             const float* __restrict__ Wih1, const float* __restrict__ Whh1,
                             const float* __restrict__ bih1, const float* __restrict__ bhh1,
                             const float* __restrict__ Wc1,  const float* __restrict__ bc1,
                             const float* __restrict__ Wc2,  const float* __restrict__ bc2,
                             float* __restrict__ out) {
  __shared__ __align__(16) _Float16 xbuf[BT * I_DIM];    // x_t tile, f16
  __shared__ __align__(16) _Float16 h0buf[BT * H_DIM];   // layer-0 h state, f16
  __shared__ __align__(16) _Float16 h1buf[BT * H_DIM];   // layer-1 h state, f16
  __shared__ __align__(16) float    gbuf[BT * G_DIM];    // gate pre-activations, f32
  __shared__ float b0s[G_DIM], b1s[G_DIM];               // bih+bhh per layer
  __shared__ float hid[BT * 32];                         // classifier hidden

  const int tid  = threadIdx.x;
  const int wave = tid >> 5;
  const int lane = tid & 31;
  const int b0r  = blockIdx.x * BT;          // batch-tile base row

  // ---- one-time setup: combined biases, zero h state ----
  if (tid < G_DIM) {
    b0s[tid] = bih0[tid] + bhh0[tid];
    b1s[tid] = bih1[tid] + bhh1[tid];
  }
#pragma unroll
  for (int e = 0; e < 4; ++e) {
    h0buf[tid + NTHREADS * e] = (_Float16)0.0f;
    h1buf[tid + NTHREADS * e] = (_Float16)0.0f;
  }

  // ---- pin this wave's weight fragments in VGPRs (loop-invariant) ----
  const int n0a = wave * 32;       // each of 8 waves owns 32 gate columns (2 N-tiles)
  const int n0b = n0a + 16;
  v16h Bih0_a  = load_B_gmem_f32(Wih0, I_DIM, n0a, 0);
  v16h Bih0_b  = load_B_gmem_f32(Wih0, I_DIM, n0b, 0);
  v16h Bhh0_a0 = load_B_gmem_f32(Whh0, H_DIM, n0a, 0);
  v16h Bhh0_a1 = load_B_gmem_f32(Whh0, H_DIM, n0a, 32);
  v16h Bhh0_b0 = load_B_gmem_f32(Whh0, H_DIM, n0b, 0);
  v16h Bhh0_b1 = load_B_gmem_f32(Whh0, H_DIM, n0b, 32);
  v16h Bih1_a0 = load_B_gmem_f32(Wih1, H_DIM, n0a, 0);
  v16h Bih1_a1 = load_B_gmem_f32(Wih1, H_DIM, n0a, 32);
  v16h Bih1_b0 = load_B_gmem_f32(Wih1, H_DIM, n0b, 0);
  v16h Bih1_b1 = load_B_gmem_f32(Wih1, H_DIM, n0b, 32);
  v16h Bhh1_a0 = load_B_gmem_f32(Whh1, H_DIM, n0a, 0);
  v16h Bhh1_a1 = load_B_gmem_f32(Whh1, H_DIM, n0a, 32);
  v16h Bhh1_b0 = load_B_gmem_f32(Whh1, H_DIM, n0b, 0);
  v16h Bhh1_b1 = load_B_gmem_f32(Whh1, H_DIM, n0b, 32);

  // c-state: fixed thread->(m,j) ownership, kept in registers across all timesteps
  float c0[4] = {0.f, 0.f, 0.f, 0.f};
  float c1[4] = {0.f, 0.f, 0.f, 0.f};

  const int dn = lane & 15, dhi = lane >> 4;  // D-matrix column / row-half

  for (int t = 0; t < T_STEPS; ++t) {
    // ---- stage x[:, t, :] tile into LDS as f16 ----
#pragma unroll
    for (int e = 0; e < 2; ++e) {
      int idx = tid + NTHREADS * e;  // 0..511 = 16 rows x 32 cols
      int m = idx >> 5, i = idx & 31;
      xbuf[idx] = (_Float16)x[(size_t)(b0r + m) * (T_STEPS * I_DIM) +
                              (size_t)t * I_DIM + i];
    }
    if (t + 1 < T_STEPS) {  // prefetch next timestep tile -> global_prefetch_b8
      const float* pf = x + (size_t)(b0r + (tid & 15)) * (T_STEPS * I_DIM) +
                        (size_t)(t + 1) * I_DIM + ((tid >> 4) & 1) * 16;
      __builtin_prefetch(pf, 0, 0);
    }
    __syncthreads();  // xbuf ready; prev-step h writes & gbuf reads ordered

    // ---- layer 0: gates = x_t @ Wih0^T + h0 @ Whh0^T ----
    {
      v16h Ax  = load_A_lds(xbuf,  I_DIM, 0);
      v16h Ah0 = load_A_lds(h0buf, H_DIM, 0);
      v16h Ah1 = load_A_lds(h0buf, H_DIM, 32);
      v8f acca = {}, accb = {};
      acca = WMMA_F16(Ax,  Bih0_a,  acca);
      acca = WMMA_F16(Ah0, Bhh0_a0, acca);
      acca = WMMA_F16(Ah1, Bhh0_a1, acca);
      accb = WMMA_F16(Ax,  Bih0_b,  accb);
      accb = WMMA_F16(Ah0, Bhh0_b0, accb);
      accb = WMMA_F16(Ah1, Bhh0_b1, accb);
#pragma unroll
      for (int r = 0; r < 8; ++r) {
        int m = r + 8 * dhi;
        gbuf[m * G_DIM + n0a + dn] = acca[r];
        gbuf[m * G_DIM + n0b + dn] = accb[r];
      }
    }
    __syncthreads();  // gates ready

    // ---- layer 0 elementwise: c0/h0 update ----
#pragma unroll
    for (int e = 0; e < 4; ++e) {
      int idx = (tid << 2) | e;       // 0..1023 = 16 rows x 64 hidden
      int m = idx >> 6, j = idx & 63;
      float gi = gbuf[m * G_DIM +       j] + b0s[      j];
      float gf = gbuf[m * G_DIM +  64 + j] + b0s[ 64 + j];
      float gg = gbuf[m * G_DIM + 128 + j] + b0s[128 + j];
      float go = gbuf[m * G_DIM + 192 + j] + b0s[192 + j];
      c0[e] = sigf(gf) * c0[e] + sigf(gi) * tanh_fast(gg);
      h0buf[m * H_DIM + j] = (_Float16)(sigf(go) * tanh_fast(c0[e]));
    }
    __syncthreads();  // h0 ready

    // ---- layer 1: gates = h0_t @ Wih1^T + h1 @ Whh1^T ----
    {
      v16h Ax0 = load_A_lds(h0buf, H_DIM, 0);
      v16h Ax1 = load_A_lds(h0buf, H_DIM, 32);
      v16h Ah0 = load_A_lds(h1buf, H_DIM, 0);
      v16h Ah1 = load_A_lds(h1buf, H_DIM, 32);
      v8f acca = {}, accb = {};
      acca = WMMA_F16(Ax0, Bih1_a0, acca);
      acca = WMMA_F16(Ax1, Bih1_a1, acca);
      acca = WMMA_F16(Ah0, Bhh1_a0, acca);
      acca = WMMA_F16(Ah1, Bhh1_a1, acca);
      accb = WMMA_F16(Ax0, Bih1_b0, accb);
      accb = WMMA_F16(Ax1, Bih1_b1, accb);
      accb = WMMA_F16(Ah0, Bhh1_b0, accb);
      accb = WMMA_F16(Ah1, Bhh1_b1, accb);
#pragma unroll
      for (int r = 0; r < 8; ++r) {
        int m = r + 8 * dhi;
        gbuf[m * G_DIM + n0a + dn] = acca[r];
        gbuf[m * G_DIM + n0b + dn] = accb[r];
      }
    }
    __syncthreads();  // gates ready

    // ---- layer 1 elementwise: c1/h1 update ----
#pragma unroll
    for (int e = 0; e < 4; ++e) {
      int idx = (tid << 2) | e;
      int m = idx >> 6, j = idx & 63;
      float gi = gbuf[m * G_DIM +       j] + b1s[      j];
      float gf = gbuf[m * G_DIM +  64 + j] + b1s[ 64 + j];
      float gg = gbuf[m * G_DIM + 128 + j] + b1s[128 + j];
      float go = gbuf[m * G_DIM + 192 + j] + b1s[192 + j];
      c1[e] = sigf(gf) * c1[e] + sigf(gi) * tanh_fast(gg);
      h1buf[m * H_DIM + j] = (_Float16)(sigf(go) * tanh_fast(c1[e]));
    }
    // next-iteration barrier (after xbuf staging) orders these writes
  }
  __syncthreads();  // final h1 ready

  // ---- classifier: hidden = relu(h1 @ Wc1^T + bc1) ; out = hidden @ Wc2^T + bc2 ----
#pragma unroll
  for (int e = 0; e < 2; ++e) {
    int idx = tid + NTHREADS * e;  // 0..511 = 16 rows x 32 units
    int m = idx >> 5, u = idx & 31;
    float s = bc1[u];
#pragma unroll 8
    for (int k = 0; k < H_DIM; ++k)
      s += (float)h1buf[m * H_DIM + k] * Wc1[u * H_DIM + k];
    hid[m * 32 + u] = fmaxf(s, 0.0f);
  }
  __syncthreads();
  if (tid < BT) {
    float s = bc2[0];
#pragma unroll 8
    for (int u = 0; u < 32; ++u) s += hid[tid * 32 + u] * Wc2[u];
    out[b0r + tid] = s;
  }
}

extern "C" void kernel_launch(void* const* d_in, const int* in_sizes, int n_in,
                              void* d_out, int out_size, void* d_ws, size_t ws_size,
                              hipStream_t stream) {
  (void)in_sizes; (void)n_in; (void)d_ws; (void)ws_size; (void)out_size;
  lstm2_fused_wmma_kernel<<<BATCH / BT, NTHREADS, 0, stream>>>(
      (const float*)d_in[0],                            // x
      (const float*)d_in[1], (const float*)d_in[2],     // Wih0, Whh0
      (const float*)d_in[3], (const float*)d_in[4],     // bih0, bhh0
      (const float*)d_in[5], (const float*)d_in[6],     // Wih1, Whh1
      (const float*)d_in[7], (const float*)d_in[8],     // bih1, bhh1
      (const float*)d_in[9], (const float*)d_in[10],    // Wc1, bc1
      (const float*)d_in[11], (const float*)d_in[12],   // Wc2, bc2
      (float*)d_out);
}
